// FactoredQuantizer_46213848105941
// MI455X (gfx1250) — matrix operand order repaired and checked by reading, not compiled, not run on hardware
//
#include <hip/hip_runtime.h>

// FactoredQuantizer for MI455X (gfx1250, wave32, WMMA).
// B=8192, M=16, N=256, C=64.
// dist(b,m,n) = ||x||^2 - 2 x.c + ||c||^2 ; argmin over n only needs c2 - 2*x.c.
// A = (-2x) in f16, B = f16 codebook slice (LDS, padded rows), f32 accumulate via
// v_wmma_f32_16x16x32_f16; c2 added post-WMMA; exact argmin via ds_swizzle butterflies.

typedef __attribute__((ext_vector_type(16))) _Float16 v16h;
typedef __attribute__((ext_vector_type(8)))  float    v8f;

#define B_TOT 8192
#define M_F   16
#define N_CB  256
#define C_DIM 64
#define WAVES 8          // waves per block; block = 256 threads
#define CB_STRIDE 72     // padded LDS row stride (halfs): 36 dwords -> conflict-free b128

// xor-butterfly min-reduce via ds_swizzle (group-of-32 mode: and=0x1F, or=0, xor=MASK)
template <int MASK>
__device__ __forceinline__ void reduce_min_f(float (&v)[8]) {
#pragma unroll
    for (int r = 0; r < 8; ++r) {
        const float o = __builtin_bit_cast(float,
            __builtin_amdgcn_ds_swizzle(__builtin_bit_cast(int, v[r]),
                                        0x1F | (MASK << 10)));
        v[r] = fminf(v[r], o);
    }
}
template <int MASK>
__device__ __forceinline__ void reduce_min_i(int (&v)[8]) {
#pragma unroll
    for (int r = 0; r < 8; ++r) {
        const int o = __builtin_amdgcn_ds_swizzle(v[r], 0x1F | (MASK << 10));
        v[r] = (o < v[r]) ? o : v[r];
    }
}

__global__ __launch_bounds__(256)
void fq_wmma_kernel(const float* __restrict__ inp,   // [B, M, C] f32
                    const float* __restrict__ cb,    // [M, N, C] f32
                    float* __restrict__ codes,       // [B, M, C] f32
                    float* __restrict__ idx_out)     // [B, M] (as float)
{
    __shared__ _Float16 s_cb[N_CB * CB_STRIDE];   // ~36 KB f16 codebook slice for this m
    __shared__ float    s_c2[N_CB];               // ||c||^2 per codeword
    __shared__ int      s_idx[WAVES][16];         // winning index per row per wave

    const int m    = blockIdx.y;
    const int t    = threadIdx.x;
    const int wave = t >> 5;
    const int lane = t & 31;
    const int nloc = lane & 15;
    const int b0   = (blockIdx.x * WAVES + wave) * 16;

    // Prefetch this wave's input tile while the block stages the codebook.
    {
        const float* pf = inp + (((size_t)(b0 + nloc)) * M_F + m) * C_DIM;
        __builtin_prefetch(pf, 0, 0);
    }

    // ---- Stage codebook slice m: f32 -> f16 into LDS (padded rows), plus ||c||^2 ----
    {
        const float* src = cb + ((size_t)m * N_CB + t) * C_DIM;
        float c2 = 0.0f;
#pragma unroll
        for (int c = 0; c < C_DIM; c += 4) {
            float4 v = *(const float4*)(src + c);
            c2 += v.x * v.x + v.y * v.y + v.z * v.z + v.w * v.w;
            s_cb[t * CB_STRIDE + c + 0] = (_Float16)v.x;
            s_cb[t * CB_STRIDE + c + 1] = (_Float16)v.y;
            s_cb[t * CB_STRIDE + c + 2] = (_Float16)v.z;
            s_cb[t * CB_STRIDE + c + 3] = (_Float16)v.w;
        }
        s_c2[t] = c2;
    }
    __syncthreads();

    // ---- Load A fragments (16x64, two K=32 steps), folding the -2 scale in ----
    // CDNA5 16-bit A layout: lane&15 = row; lanes 0-15 hold K {0-7,16-23},
    // lanes 16-31 hold K {8-15,24-31}. Per lane each 8-half group is K-contiguous,
    // so each group is one float4-pair load from global.
    v16h a0, a1;
    {
        const int khalfA = (lane >> 4) * 8;
        const float* arow = inp + (((size_t)(b0 + nloc)) * M_F + m) * C_DIM + khalfA;
#pragma unroll
        for (int g = 0; g < 2; ++g) {               // half-groups: K+0 / K+16
            float4 p0 = *(const float4*)(arow + 16 * g);
            float4 p1 = *(const float4*)(arow + 16 * g + 4);
            float4 q0 = *(const float4*)(arow + 16 * g + 32);
            float4 q1 = *(const float4*)(arow + 16 * g + 36);
            a0[8 * g + 0] = (_Float16)(-2.0f * p0.x);
            a0[8 * g + 1] = (_Float16)(-2.0f * p0.y);
            a0[8 * g + 2] = (_Float16)(-2.0f * p0.z);
            a0[8 * g + 3] = (_Float16)(-2.0f * p0.w);
            a0[8 * g + 4] = (_Float16)(-2.0f * p1.x);
            a0[8 * g + 5] = (_Float16)(-2.0f * p1.y);
            a0[8 * g + 6] = (_Float16)(-2.0f * p1.z);
            a0[8 * g + 7] = (_Float16)(-2.0f * p1.w);
            a1[8 * g + 0] = (_Float16)(-2.0f * q0.x);
            a1[8 * g + 1] = (_Float16)(-2.0f * q0.y);
            a1[8 * g + 2] = (_Float16)(-2.0f * q0.z);
            a1[8 * g + 3] = (_Float16)(-2.0f * q0.w);
            a1[8 * g + 4] = (_Float16)(-2.0f * q1.x);
            a1[8 * g + 5] = (_Float16)(-2.0f * q1.y);
            a1[8 * g + 6] = (_Float16)(-2.0f * q1.z);
            a1[8 * g + 7] = (_Float16)(-2.0f * q1.w);
        }
    }

    // Preload all 16 c2 values this lane needs (n = nloc + 16*t16) -> fixed VGPRs.
    float c2arr[16];
#pragma unroll
    for (int i = 0; i < 16; ++i) c2arr[i] = s_c2[i * 16 + nloc];

    float minv[8];
    int   mini[8];
#pragma unroll
    for (int r = 0; r < 8; ++r) { minv[r] = 3.4e38f; mini[r] = 0; }

    // B layout (dense 32x16): lane&15 = column N;
    // lanes 0-15 hold K {0-7,16-23}, lanes 16-31 hold K {8-15,24-31} (matches A).
    const int kbB = (lane >> 4) * 8;

#pragma unroll
    for (int t16 = 0; t16 < 16; ++t16) {
        const int n = t16 * 16 + nloc;

        const uint4* bp = (const uint4*)&s_cb[n * CB_STRIDE + kbB];
        v16h bf0, bf1;
        ((uint4*)&bf0)[0] = bp[0];   // K-step 0: halfs 0..7
        ((uint4*)&bf0)[1] = bp[2];   //           halfs 16..23
        ((uint4*)&bf1)[0] = bp[4];   // K-step 1: +32 halfs
        ((uint4*)&bf1)[1] = bp[6];

        v8f acc = {};  // SRC2 = inline 0: WMMA chain starts with no VALU dependency
        acc = __builtin_amdgcn_wmma_f32_16x16x32_f16(
                  false, a0, false, bf0, (short)0, acc, false, false);
        acc = __builtin_amdgcn_wmma_f32_16x16x32_f16(
                  false, a1, false, bf1, (short)0, acc, false, false);

        const float c2n = c2arr[t16];
#pragma unroll
        for (int r = 0; r < 8; ++r) {
            const float d = c2n + acc[r];
            if (d < minv[r]) { minv[r] = d; mini[r] = n; }
        }
    }

    // ---- Exact cross-lane argmin (16 lanes share a row group) ----
    // Phase 1: min-reduce the value.  Phase 2: min-reduce the index among exact winners
    // (== jnp.argmin first-occurrence: smallest n attaining the min).
    float vloc[8];
#pragma unroll
    for (int r = 0; r < 8; ++r) vloc[r] = minv[r];
    reduce_min_f<1>(minv);  reduce_min_f<2>(minv);
    reduce_min_f<4>(minv);  reduce_min_f<8>(minv);
#pragma unroll
    for (int r = 0; r < 8; ++r) mini[r] = (vloc[r] == minv[r]) ? mini[r] : 0x7FFFFFFF;
    reduce_min_i<1>(mini);  reduce_min_i<2>(mini);
    reduce_min_i<4>(mini);  reduce_min_i<8>(mini);

    if (nloc == 0) {
        const int rb = (lane >> 4) * 8;   // lane0 -> rows 0-7, lane16 -> rows 8-15
#pragma unroll
        for (int r = 0; r < 8; ++r) s_idx[wave][rb + r] = mini[r];
    }
    __syncthreads();

    // ---- idx output (float-typed d_out tail) ----
    if (lane < 16) {
        idx_out[(size_t)(b0 + lane) * M_F + m] = (float)s_idx[wave][lane];
    }

    // ---- Gather codes: 16 rows x 64 f32 from the L2-resident codebook ----
#pragma unroll
    for (int r = 0; r < 16; ++r) {
        const int n = s_idx[wave][r];
        const float2 val =
            *(const float2*)(cb + (((size_t)m * N_CB + n) * C_DIM) + lane * 2);
        *(float2*)(codes + (((size_t)(b0 + r) * M_F + m) * C_DIM) + lane * 2) = val;
    }
}

extern "C" void kernel_launch(void* const* d_in, const int* in_sizes, int n_in,
                              void* d_out, int out_size, void* d_ws, size_t ws_size,
                              hipStream_t stream) {
    const float* inp = (const float*)d_in[0];   // inputs   [B, M, C] f32
    const float* cb  = (const float*)d_in[1];   // codebook [M, N, C] f32
    float* codes   = (float*)d_out;                                   // [B,M,C]
    float* idx_out = (float*)d_out + (size_t)B_TOT * M_F * C_DIM;     // [B,M]

    dim3 grid(B_TOT / (16 * WAVES), M_F);
    fq_wmma_kernel<<<grid, 256, 0, stream>>>(inp, cb, codes, idx_out);
}
